// ContextualAttention_Enhance_51221779972109
// MI455X (gfx1250) — matrix operand order repaired and checked by reading, not compiled, not run on hardware
//
#include <hip/hip_runtime.h>
#include <stdint.h>

// ---------------- types ----------------
typedef __attribute__((ext_vector_type(16))) __bf16 v16bf;
typedef __attribute__((ext_vector_type(8)))  __bf16 v8bf;
typedef __attribute__((ext_vector_type(8)))  float  v8f;

static constexpr int BQ   = 2;
static constexpr int CIN  = 64;
static constexpr int HH   = 128;
static constexpr int WW_  = 128;
static constexpr int IC   = 16;
static constexpr int KS   = 7;
static constexpr int NQH  = 32;
static constexpr int NQW  = 32;
static constexpr int NQ   = NQH * NQW;     // 1024 query patches
static constexpr int NKEY = HH * WW_;      // 16384 key/value patches
static constexpr int NGRP = KS * KS;       // 49 offset groups
static constexpr int FPAD = 832;           // 52*16 padded feature dim (52 = 4 waves x 13 tiles)
static constexpr int NCH  = FPAD / 32;     // 26 K-chunks for S phase
static constexpr int NT   = 13;            // feature tiles per Z-wave (compile-time!)
static constexpr float SCALE_F = 10.0f;

__device__ __forceinline__ uint16_t f2bf(float f) {
  union { float f; uint32_t u; } x; x.f = f;
  uint32_t r = x.u + 0x7FFFu + ((x.u >> 16) & 1u);  // round-to-nearest-even
  return (uint16_t)(r >> 16);
}

__device__ __forceinline__ v16bf make16(v8bf lo, v8bf hi) {
  union { v16bf w; v8bf h[2]; } u;
  u.h[0] = lo; u.h[1] = hi;
  return u.w;
}

// -------- kernel 1: fused 1x1 convs -> bf16 maps (Q from Wg, V from Wt, K from Wp)
__global__ void k_proj(const float* __restrict__ xin,
                       const float* __restrict__ Wg, const float* __restrict__ bg,
                       const float* __restrict__ Wt, const float* __restrict__ bt,
                       const float* __restrict__ Wp, const float* __restrict__ bp,
                       uint16_t* __restrict__ Qm, uint16_t* __restrict__ Km,
                       uint16_t* __restrict__ Vm) {
  __shared__ float sWg[IC * CIN], sWt[IC * CIN], sWp[IC * CIN];
  __shared__ float sbg[IC], sbt[IC], sbp[IC];
  for (int i = threadIdx.x; i < IC * CIN; i += blockDim.x) {
    sWg[i] = Wg[i]; sWt[i] = Wt[i]; sWp[i] = Wp[i];
  }
  if (threadIdx.x < IC) {
    sbg[threadIdx.x] = bg[threadIdx.x];
    sbt[threadIdx.x] = bt[threadIdx.x];
    sbp[threadIdx.x] = bp[threadIdx.x];
  }
  __syncthreads();
  int gid = blockIdx.x * blockDim.x + threadIdx.x;     // over BQ*NKEY
  if (gid >= BQ * NKEY) return;
  int bi = gid / NKEY, p = gid % NKEY;
  float x[CIN];
  const float* xp = xin + (size_t)bi * CIN * NKEY + p;
#pragma unroll
  for (int c = 0; c < CIN; ++c) x[c] = xp[(size_t)c * NKEY];
  for (int o = 0; o < IC; ++o) {
    float ag = sbg[o], at = sbt[o], ap = sbp[o];
#pragma unroll
    for (int c = 0; c < CIN; ++c) {
      float xv = x[c];
      ag += sWg[o * CIN + c] * xv;
      at += sWt[o * CIN + c] * xv;
      ap += sWp[o * CIN + c] * xv;
    }
    size_t idx = ((size_t)bi * IC + o) * NKEY + p;
    Qm[idx] = f2bf(ag); Vm[idx] = f2bf(at); Km[idx] = f2bf(ap);
  }
}

// -------- kernel 2: query patches Qp[b][l][f'] (f' = (dy*7+dx)*16 + c, padded to 832)
__global__ void k_patchq(const uint16_t* __restrict__ Qm, uint16_t* __restrict__ Qp) {
  int gid = blockIdx.x * blockDim.x + threadIdx.x;     // BQ*NQ*FPAD
  if (gid >= BQ * NQ * FPAD) return;
  int f = gid % FPAD; int rem = gid / FPAD;
  int l = rem % NQ;   int bi = rem / NQ;
  int g = f >> 4, c = f & 15;
  uint16_t v = 0;
  if (g < NGRP) {
    int dy = g / KS, dx = g % KS;
    int qh = l >> 5, qw = l & 31;
    int r = qh * 4 + dy - 1, col = qw * 4 + dx - 1;    // pad (1,2,1,2) semantics
    if ((unsigned)r < HH && (unsigned)col < WW_)
      v = Qm[((size_t)bi * IC + c) * NKEY + r * WW_ + col];
  }
  Qp[gid] = v;
}

// -------- kernel 3a: key patches Kp[b][m][f'] (row-major, f fastest -> coalesced)
__global__ void k_patchk(const uint16_t* __restrict__ Km, uint16_t* __restrict__ Kp) {
  int gid = blockIdx.x * blockDim.x + threadIdx.x;     // BQ*NKEY*FPAD
  if (gid >= BQ * NKEY * FPAD) return;
  int f = gid % FPAD; int rem = gid / FPAD;
  int m = rem % NKEY; int bi = rem / NKEY;
  int g = f >> 4, c = f & 15;
  uint16_t v = 0;
  if (g < NGRP) {
    int dy = g / KS, dx = g % KS;
    int my = m >> 7, mx = m & 127;
    int r = my + dy - 3, col = mx + dx - 3;            // pad 3 each side
    if ((unsigned)r < HH && (unsigned)col < WW_)
      v = Km[((size_t)bi * IC + c) * NKEY + r * WW_ + col];
  }
  Kp[gid] = v;
}

// -------- kernel 3b: value patches, transposed: Vt[b][f'][m] (m fastest -> coalesced)
__global__ void k_patchv(const uint16_t* __restrict__ Vm, uint16_t* __restrict__ Vt) {
  int gid = blockIdx.x * blockDim.x + threadIdx.x;     // BQ*FPAD*NKEY
  if (gid >= BQ * FPAD * NKEY) return;
  int m = gid % NKEY; int rem = gid / NKEY;
  int f = rem % FPAD; int bi = rem / FPAD;
  int g = f >> 4, c = f & 15;
  uint16_t v = 0;
  if (g < NGRP) {
    int dy = g / KS, dx = g % KS;
    int my = m >> 7, mx = m & 127;
    int r = my + dy - 3, col = mx + dx - 3;
    if ((unsigned)r < HH && (unsigned)col < WW_)
      v = Vm[((size_t)bi * IC + c) * NKEY + r * WW_ + col];
  }
  Vt[gid] = v;
}

// -------- kernel 4: flash attention, z = softmax(10*Q K^T) V
// Block = 16 queries, 8 waves. Waves 0-3: S^T tiles (keys=A rows, queries=B cols),
// 26 WMMAs/iter each. Waves 4-7: 13 Z^T accumulator tiles each (A = V^T feature
// rows, B = P columns), 26 WMMAs/iter each — balanced pipeline, all loops are
// compile-time trip-count so accumulators stay in VGPRs (no scratch spills).
__global__ __launch_bounds__(256) void k_attn(const uint16_t* __restrict__ Qp,
                                              const uint16_t* __restrict__ Kp,
                                              const uint16_t* __restrict__ Vt,
                                              float* __restrict__ Z) {
  __shared__ __align__(16) uint16_t Qs[16 * FPAD];     // 26.6 KB query tile
  __shared__ __align__(16) float    Ss[64 * 16];       // S^T staging (key, query)
  __shared__ __align__(16) uint16_t Ps[16 * 64];       // P (query, key) bf16
  __shared__ float m_run[16], l_run[16], corr[16];

  const int tid  = threadIdx.x;
  const int wave = tid >> 5;
  const int lane = tid & 31;
  const int ln   = lane & 15;       // WMMA N / column index
  const int lh   = lane >> 4;       // lane-half select
  const int bi   = blockIdx.x >> 6;
  const int q0   = (blockIdx.x & 63) * 16;

  { // load Q tile into LDS (layout identical to global: [q][f'])
    const uint32_t* qsrc = (const uint32_t*)(Qp + ((size_t)bi * NQ + q0) * FPAD);
    uint32_t* qdst = (uint32_t*)Qs;
    for (int i = tid; i < 16 * FPAD / 2; i += 256) qdst[i] = qsrc[i];
  }
  if (tid < 16) { m_run[tid] = -1e30f; l_run[tid] = 0.0f; }
  __syncthreads();

  v8f acc[NT];
  const int ft0 = (wave >= 4) ? (wave - 4) * NT : 0;
#pragma unroll
  for (int j = 0; j < NT; ++j)
    acc[j] = (v8f){0.f, 0.f, 0.f, 0.f, 0.f, 0.f, 0.f, 0.f};

  for (int t = 0; t < NKEY / 64; ++t) {
    const int k0 = t * 64;
    if (wave < 4) {
      // ---- S phase: S^T(16 keys x 16 queries) = K_tile(16x32) x Q^T(32x16)
      const int key = k0 + wave * 16 + ln;
      const uint16_t* arow = Kp + ((size_t)bi * NKEY + key) * FPAD + lh * 16;
      v8f s = (v8f){0.f, 0.f, 0.f, 0.f, 0.f, 0.f, 0.f, 0.f};
#pragma unroll
      for (int kc = 0; kc < NCH; ++kc) {
        v8bf a0 = *(const v8bf*)(arow + kc * 32);
        v8bf a1 = *(const v8bf*)(arow + kc * 32 + 8);
        const uint16_t* brow = Qs + ln * FPAD + kc * 32 + lh * 16;
        v8bf b0 = *(const v8bf*)(brow);
        v8bf b1 = *(const v8bf*)(brow + 8);
        s = __builtin_amdgcn_wmma_f32_16x16x32_bf16(
                false, make16(a0, a1), false, make16(b0, b1),
                (short)0, s, false, false);
      }
#pragma unroll
      for (int r = 0; r < 8; ++r) {
        int krow = wave * 16 + r + 8 * lh;            // key row within 64-tile
        Ss[krow * 16 + ln] = s[r] * SCALE_F;
      }
      if (t + 1 < NKEY / 64)                          // L2 prefetch of next K rows
        __builtin_prefetch(Kp + ((size_t)bi * NKEY + k0 + 64 + wave * 16 + ln) * FPAD, 0, 1);
    } else {
      if (t + 1 < NKEY / 64)                          // prefetch next V^T columns
        __builtin_prefetch(Vt + ((size_t)bi * FPAD + ft0 * 16 + ln) * NKEY + k0 + 64, 0, 1);
    }
    __syncthreads();

    // ---- online softmax (16 threads, one per query column)
    if (tid < 16) {
      const int q = tid;
      float mx = m_run[q];
      for (int k2 = 0; k2 < 64; ++k2) mx = fmaxf(mx, Ss[k2 * 16 + q]);
      float c0 = __expf(m_run[q] - mx);
      float ssum = l_run[q] * c0;
      for (int k2 = 0; k2 < 64; ++k2) {
        float e = __expf(Ss[k2 * 16 + q] - mx);
        Ps[q * 64 + k2] = f2bf(e);
        ssum += e;
      }
      m_run[q] = mx; l_run[q] = ssum; corr[q] = c0;
    }
    __syncthreads();

    // ---- Z phase: Z^T(16f x 16q) += V^T(16f x 32k) x P(32k x 16q), 2 chunks
    if (wave >= 4) {
      float fct = corr[ln];
      v16bf Bf[2];
#pragma unroll
      for (int ch = 0; ch < 2; ++ch) {
        const uint16_t* pb = Ps + ln * 64 + ch * 32 + lh * 16;
        Bf[ch] = make16(*(const v8bf*)pb, *(const v8bf*)(pb + 8));
      }
#pragma unroll
      for (int j = 0; j < NT; ++j) {
        int f = (ft0 + j) * 16 + ln;
        const uint16_t* vrow = Vt + ((size_t)bi * FPAD + f) * NKEY + k0 + lh * 16;
        v8f a = acc[j];
#pragma unroll
        for (int r2 = 0; r2 < 8; ++r2) a[r2] *= fct;  // rescale: per-column factor
#pragma unroll
        for (int ch = 0; ch < 2; ++ch) {
          v8bf a0 = *(const v8bf*)(vrow + ch * 32);
          v8bf a1 = *(const v8bf*)(vrow + ch * 32 + 8);
          a = __builtin_amdgcn_wmma_f32_16x16x32_bf16(
                  false, make16(a0, a1), false, Bf[ch],
                  (short)0, a, false, false);
        }
        acc[j] = a;
      }
    }
  }
  __syncthreads();

  // ---- normalize by row sum and write z[b][q][f']
  if (wave >= 4) {
    float inv = 1.0f / l_run[ln];
    int q = q0 + ln;
#pragma unroll
    for (int j = 0; j < NT; ++j) {
#pragma unroll
      for (int r = 0; r < 8; ++r) {
        int f = (ft0 + j) * 16 + r + 8 * lh;
        Z[((size_t)bi * NQ + q) * FPAD + f] = acc[j][r] * inv;
      }
    }
  }
}

// -------- kernel 5: fold-add + count normalize + 1x1 conv (16->64) + residual
__global__ void k_out(const float* __restrict__ Z, const float* __restrict__ Ww,
                      const float* __restrict__ bw, const float* __restrict__ xin,
                      float* __restrict__ out) {
  __shared__ float sW[CIN * IC];
  __shared__ float sb[CIN];
  for (int i = threadIdx.x; i < CIN * IC; i += blockDim.x) sW[i] = Ww[i];
  for (int i = threadIdx.x; i < CIN; i += blockDim.x) sb[i] = bw[i];
  __syncthreads();
  int gid = blockIdx.x * blockDim.x + threadIdx.x;     // BQ*NKEY pixels
  if (gid >= BQ * NKEY) return;
  int bi = gid / NKEY, p = gid % NKEY;
  int y = p >> 7, x = p & 127;
  float num[IC];
#pragma unroll
  for (int i = 0; i < IC; ++i) num[i] = 0.0f;
  int cnt = 0;
  for (int dy = 0; dy < KS; ++dy) {
    int tr = y + 3 - dy;
    if (tr & 3) continue;
    int qh = tr >> 2;
    if ((unsigned)qh >= NQH) continue;
    for (int dx = 0; dx < KS; ++dx) {
      int tc = x + 3 - dx;
      if (tc & 3) continue;
      int qw = tc >> 2;
      if ((unsigned)qw >= NQW) continue;
      int l = qh * NQW + qw;
      const float* zp = Z + ((size_t)bi * NQ + l) * FPAD + (dy * KS + dx) * IC;
#pragma unroll
      for (int ic = 0; ic < IC; ++ic) num[ic] += zp[ic];
      ++cnt;
    }
  }
  float inv = 1.0f / (float)cnt;
  float v[IC];
#pragma unroll
  for (int ic = 0; ic < IC; ++ic) v[ic] = num[ic] * inv;
  const float* xb = xin + (size_t)bi * CIN * NKEY + p;
  float* ob = out + (size_t)bi * CIN * NKEY + p;
  for (int co = 0; co < CIN; ++co) {
    float a = sb[co];
#pragma unroll
    for (int ic = 0; ic < IC; ++ic) a += sW[co * IC + ic] * v[ic];
    ob[(size_t)co * NKEY] = xb[(size_t)co * NKEY] + a;
  }
}

// ---------------- launcher ----------------
extern "C" void kernel_launch(void* const* d_in, const int* in_sizes, int n_in,
                              void* d_out, int out_size, void* d_ws, size_t ws_size,
                              hipStream_t stream) {
  const float* xin = (const float*)d_in[0];
  const float* Wg  = (const float*)d_in[1];
  const float* bg  = (const float*)d_in[2];
  const float* Wt  = (const float*)d_in[3];
  const float* bt  = (const float*)d_in[4];
  const float* Wp  = (const float*)d_in[5];
  const float* bp  = (const float*)d_in[6];
  const float* Ww  = (const float*)d_in[7];
  const float* bw  = (const float*)d_in[8];
  float* out = (float*)d_out;

  char* ws = (char*)d_ws;
  const size_t szMap = (size_t)BQ * IC * NKEY * sizeof(uint16_t);   // 1 MB
  const size_t szQp  = (size_t)BQ * NQ * FPAD * sizeof(uint16_t);   // 3.4 MB
  const size_t szKp  = (size_t)BQ * NKEY * FPAD * sizeof(uint16_t); // 54.5 MB
  uint16_t* Qm = (uint16_t*)(ws);
  uint16_t* Km = (uint16_t*)(ws + szMap);
  uint16_t* Vm = (uint16_t*)(ws + 2 * szMap);
  uint16_t* Qp = (uint16_t*)(ws + 3 * szMap);
  uint16_t* Kp = (uint16_t*)(ws + 3 * szMap + szQp);
  uint16_t* Vt = (uint16_t*)(ws + 3 * szMap + szQp + szKp);
  float*    Zb = (float*)   (ws + 3 * szMap + szQp + 2 * szKp);     // 6.8 MB

  k_proj<<<(BQ * NKEY) / 128, 128, 0, stream>>>(xin, Wg, bg, Wt, bt, Wp, bp, Qm, Km, Vm);
  k_patchq<<<(BQ * NQ * FPAD) / 256, 256, 0, stream>>>(Qm, Qp);
  k_patchk<<<(BQ * NKEY * FPAD) / 256, 256, 0, stream>>>(Km, Kp);
  k_patchv<<<(BQ * FPAD * NKEY) / 256, 256, 0, stream>>>(Vm, Vt);
  k_attn<<<BQ * (NQ / 16), 256, 0, stream>>>(Qp, Kp, Vt, Zb);
  k_out<<<(BQ * NKEY) / 256, 256, 0, stream>>>(Zb, Ww, bw, xin, out);
}
